// NsaAttention_74371653697875
// MI455X (gfx1250) — matrix operand order repaired and checked by reading, not compiled, not run on hardware
//
#include <hip/hip_runtime.h>
#include <hip/hip_bf16.h>
#include <math.h>

// ---------------------------------------------------------------------------
// NSA forward for MI455X (gfx1250, wave32, WMMA f32_16x16x32_f16).
// One wave per query position; M dim of every WMMA tile = the 16 query heads.
// ---------------------------------------------------------------------------

typedef __attribute__((ext_vector_type(16))) _Float16 v16h;
typedef __attribute__((ext_vector_type(8)))  _Float16 v8h;
typedef __attribute__((ext_vector_type(8)))  float    v8f;

#define S_LEN   2048
#define HQ      16
#define DK      128
#define DV      128
#define T_CMP   127     // (2048-32)/16 + 1
#define NBLK    32      // 2048/64
#define TOPN    8
#define SELW    64
#define WIN     512
#define BIGF    1e30f

#define WMMA_F16(A, B, C) \
  __builtin_amdgcn_wmma_f32_16x16x32_f16(false, (A), false, (B), (short)0, (C), false, false)

__device__ __forceinline__ float warp16_max(float x) {
  x = fmaxf(x, __shfl_xor(x, 1, 32));
  x = fmaxf(x, __shfl_xor(x, 2, 32));
  x = fmaxf(x, __shfl_xor(x, 4, 32));
  x = fmaxf(x, __shfl_xor(x, 8, 32));
  return x;
}
__device__ __forceinline__ float warp16_sum(float x) {
  x += __shfl_xor(x, 1, 32);
  x += __shfl_xor(x, 2, 32);
  x += __shfl_xor(x, 4, 32);
  x += __shfl_xor(x, 8, 32);
  return x;
}

struct Flash {
  v8f   acc[8];   // 8 dv-column tiles of the 16(head)x16(dv) f32 C layout
  float m[8];     // row stats; lane half selects rows r (+8)
  float l[8];
  __device__ __forceinline__ void reset() {
    #pragma unroll
    for (int i = 0; i < 8; ++i) { v8f z = {}; acc[i] = z; m[i] = -BIGF; l[i] = 0.f; }
  }
};

// One 32-key flash step: scores (8 WMMA) + online softmax + PV (8 WMMA).
// Tokens are tbase..tbase+31; a token is valid iff lo <= tok <= hi.
__device__ __forceinline__ void attn_group32(
    Flash& st, const v16h* qa, const _Float16* kh, const _Float16* vh,
    int tbase, int lo, int hi, int clampMax,
    _Float16* vstage, _Float16* pstage, int lane)
{
  const int half = lane >> 4;
  const int ln   = lane & 15;

  __syncthreads();                       // previous group done reading LDS
  // cooperative V stage: 32 tokens x 128 halfs (256B per lane, vectorized)
  {
    int tok = tbase + lane;
    tok = tok < 0 ? 0 : (tok > clampMax ? clampMax : tok);
    const v16h* src = (const v16h*)(vh + (size_t)tok * DV);
    v16h* dst = (v16h*)(vstage + lane * DV);
    #pragma unroll
    for (int c = 0; c < 8; ++c) dst[c] = src[c];
  }

  // scores: Q(16x128) x K^T -> two 16x16 tiles (keys tbase+0..15 / +16..31)
  v8f s0 = {}, s1 = {};
  int t0 = tbase + ln;
  int t1 = t0 + 16;
  int ct0 = t0 < 0 ? 0 : (t0 > clampMax ? clampMax : t0);
  int ct1 = t1 < 0 ? 0 : (t1 > clampMax ? clampMax : t1);
  #pragma unroll
  for (int c = 0; c < 4; ++c) {
    // B layout: lane -> key column, element e -> dk index c*32 + e + half*16
    v16h kb0 = *(const v16h*)(kh + (size_t)ct0 * DK + c * 32 + half * 16);
    v16h kb1 = *(const v16h*)(kh + (size_t)ct1 * DK + c * 32 + half * 16);
    s0 = WMMA_F16(qa[c], kb0, s0);
    s1 = WMMA_F16(qa[c], kb1, s1);
  }

  const bool v0 = (t0 >= lo) && (t0 <= hi);
  const bool v1 = (t1 >= lo) && (t1 <= hi);
  #pragma unroll
  for (int r = 0; r < 8; ++r) {
    float a0 = v0 ? s0[r] : -BIGF;
    float a1 = v1 ? s1[r] : -BIGF;
    float mx   = warp16_max(fmaxf(a0, a1));
    float mnew = fmaxf(st.m[r], mx);
    float alpha = __expf(st.m[r] - mnew);
    float p0 = v0 ? __expf(a0 - mnew) : 0.f;
    float p1 = v1 ? __expf(a1 - mnew) : 0.f;
    float ps = warp16_sum(p0 + p1);
    st.l[r] = st.l[r] * alpha + ps;
    st.m[r] = mnew;
    #pragma unroll
    for (int nc = 0; nc < 8; ++nc) st.acc[nc][r] *= alpha;
    int row = r + half * 8;
    pstage[row * 32 + ln]      = (_Float16)p0;   // f16 so pa reload vectorizes
    pstage[row * 32 + 16 + ln] = (_Float16)p1;
  }
  __syncthreads();                       // pstage + vstage visible

  // P as A fragment (16 heads x 32 keys): two aligned 16B LDS loads per lane.
  // A layout: e=0..7 -> kk = e + half*8 (contig), e=8..15 -> kk = e+8+half*8.
  v8h plo = *(const v8h*)(pstage + ln * 32 + half * 8);
  v8h phi = *(const v8h*)(pstage + ln * 32 + 16 + half * 8);
  v16h pa = __builtin_shufflevector(plo, phi,
                                    0, 1, 2, 3, 4, 5, 6, 7,
                                    8, 9, 10, 11, 12, 13, 14, 15);
  // PV: 8 dv tiles of 16
  #pragma unroll
  for (int nc = 0; nc < 8; ++nc) {
    v16h vb;
    #pragma unroll
    for (int e = 0; e < 16; ++e)
      vb[e] = vstage[(e + ((e >= 8) ? 8 : 0) + half * 8) * DV + nc * 16 + ln];
    st.acc[nc] = WMMA_F16(pa, vb, st.acc[nc]);
  }
}

// Prep: fp32 -> f16 copies of K/V, and pooled cmp_k/cmp_v (also f16).
extern "C" __global__ void nsa_prep(const float* __restrict__ k,
                                    const float* __restrict__ v,
                                    const float* __restrict__ wk,
                                    const float* __restrict__ wv,
                                    _Float16* kh, _Float16* vh,
                                    _Float16* cmpkh, _Float16* cmpvh)
{
  int i = blockIdx.x * blockDim.x + threadIdx.x;
  if (i < S_LEN * DK) {
    kh[i] = (_Float16)k[i];
    vh[i] = (_Float16)v[i];
  }
  if (i < T_CMP * DK) {
    int t = i / DK, d = i % DK;
    float ak = 0.f, av = 0.f;
    #pragma unroll
    for (int l = 0; l < 32; ++l) {
      ak += k[(t * 16 + l) * DK + d] * wk[l];
      av += v[(t * 16 + l) * DK + d] * wv[l];
    }
    cmpkh[i] = (_Float16)ak;
    cmpvh[i] = (_Float16)av;
  }
}

extern "C" __global__ void __launch_bounds__(32)
nsa_main(const float* __restrict__ q,
         const _Float16* __restrict__ kh, const _Float16* __restrict__ vh,
         const _Float16* __restrict__ cmpkh, const _Float16* __restrict__ cmpvh,
         const float* __restrict__ gate_w, const float* __restrict__ gate_b,
         float* __restrict__ out)
{
  __shared__ float sc[HQ * 128];                 // compressed scores/probs
  __shared__ __align__(16) float outbuf[HQ * DV]; // gated output accumulator
  __shared__ __align__(16) _Float16 pstage[16 * 32]; // P relayout bounce (f16)
  __shared__ float pgrp[128];
  __shared__ float gates[HQ * 3];
  __shared__ int   selidx[TOPN];
  __shared__ int   seldup[TOPN];
  __shared__ __align__(32) _Float16 vstage[32 * DV];

  const int s    = blockIdx.x;
  const int lane = threadIdx.x;
  const int half = lane >> 4;
  const int ln   = lane & 15;
  const float scale = 0.088388347648318447f;     // 128^-0.5 folded into q

  // Q A-fragments: rows = 16 heads, 4 chunks of dk
  v16h qa[4];
  #pragma unroll
  for (int c = 0; c < 4; ++c)
    #pragma unroll
    for (int e = 0; e < 16; ++e) {
      int kk = e + half * 16;
      qa[c][e] = (_Float16)(q[((size_t)s * HQ + ln) * DK + c * 32 + kk] * scale);
    }

  // gates = sigmoid(q @ gate_w + gate_b), one head per lane (halves split d)
  #pragma unroll
  for (int i = 0; i < 3; ++i) {
    float p = 0.f;
    for (int d = half * 64; d < half * 64 + 64; ++d)
      p += q[((size_t)s * HQ + ln) * DK + d] * gate_w[d * 3 + i];
    p += __shfl_xor(p, 16, 32);
    if (half == 0) gates[ln * 3 + i] = 1.f / (1.f + __expf(-(p + gate_b[i])));
  }

  // ---------------- compressed attention: masked scores into LDS ----------
  for (int g = 0; g < 4; ++g) {
    v8f s0 = {}, s1 = {};
    int t0 = g * 32 + ln, t1 = t0 + 16;
    int ct0 = t0 > T_CMP - 1 ? T_CMP - 1 : t0;
    int ct1 = t1 > T_CMP - 1 ? T_CMP - 1 : t1;
    #pragma unroll
    for (int c = 0; c < 4; ++c) {
      v16h kb0 = *(const v16h*)(cmpkh + (size_t)ct0 * DK + c * 32 + half * 16);
      v16h kb1 = *(const v16h*)(cmpkh + (size_t)ct1 * DK + c * 32 + half * 16);
      s0 = WMMA_F16(qa[c], kb0, s0);
      s1 = WMMA_F16(qa[c], kb1, s1);
    }
    bool m0 = (t0 < T_CMP) && (s >= t0 * 16 + 31);   // cmask: pos >= blk_end
    bool m1 = (t1 < T_CMP) && (s >= t1 * 16 + 31);
    #pragma unroll
    for (int r = 0; r < 8; ++r) {
      int row = r + half * 8;
      sc[row * 128 + t0] = m0 ? s0[r] : -BIGF;
      sc[row * 128 + t1] = m1 ? s1[r] : -BIGF;
    }
  }
  __syncthreads();

  // exact softmax per head (128 cols resident in LDS)
  {
    float mloc = -BIGF;
    for (int t = half * 64; t < half * 64 + 64; ++t)
      mloc = fmaxf(mloc, sc[ln * 128 + t]);
    mloc = fmaxf(mloc, __shfl_xor(mloc, 16, 32));
    float lloc = 0.f;
    for (int t = half * 64; t < half * 64 + 64; ++t) {
      float sv = sc[ln * 128 + t];
      float e  = (sv > -0.5f * BIGF) ? __expf(sv - mloc) : 0.f;
      sc[ln * 128 + t] = e;
      lloc += e;
    }
    lloc += __shfl_xor(lloc, 16, 32);
    float inv = 1.f / fmaxf(lloc, 1e-20f);
    for (int t = half * 64; t < half * 64 + 64; ++t)
      sc[ln * 128 + t] *= inv;
  }
  __syncthreads();

  // p_grp[t] = sum over heads of p_cmp
  for (int t = lane; t < 128; t += 32) {
    float a = 0.f;
    #pragma unroll
    for (int h = 0; h < HQ; ++h) a += sc[h * 128 + t];
    pgrp[t] = a;
  }
  __syncthreads();

  // cmp_o = P @ cmp_v via WMMA (P already normalized in LDS)
  {
    v8f cacc[8];
    #pragma unroll
    for (int i = 0; i < 8; ++i) { v8f z = {}; cacc[i] = z; }
    for (int g = 0; g < 4; ++g) {
      __syncthreads();
      {
        int tok = g * 32 + lane;
        tok = tok > T_CMP - 1 ? T_CMP - 1 : tok;
        const v16h* src = (const v16h*)(cmpvh + (size_t)tok * DV);
        v16h* dst = (v16h*)(vstage + lane * DV);
        #pragma unroll
        for (int c = 0; c < 8; ++c) dst[c] = src[c];
      }
      __syncthreads();
      v16h pa;
      #pragma unroll
      for (int e = 0; e < 16; ++e) {
        int kk = e + ((e >= 8) ? 8 : 0) + half * 8;
        pa[e] = (_Float16)sc[ln * 128 + g * 32 + kk];
      }
      #pragma unroll
      for (int nc = 0; nc < 8; ++nc) {
        v16h vb;
        #pragma unroll
        for (int e = 0; e < 16; ++e)
          vb[e] = vstage[(e + ((e >= 8) ? 8 : 0) + half * 8) * DV + nc * 16 + ln];
        cacc[nc] = WMMA_F16(pa, vb, cacc[nc]);
      }
    }
    #pragma unroll
    for (int nc = 0; nc < 8; ++nc)
      #pragma unroll
      for (int r = 0; r < 8; ++r) {
        int h = r + half * 8;
        outbuf[h * DV + nc * 16 + ln] = gates[h * 3 + 0] * cacc[nc][r];
      }
  }

  // ---------------- block selection (top-8 of 32, jax.lax.top_k order) ----
  {
    int j = lane;                                  // one block per lane
    float pv = 0.f;
    for (int t = 4 * j - 1; t <= 4 * j + 3; ++t)
      if (t >= 0 && t < T_CMP &&
          (t * 16 < (j + 1) * SELW) && (t * 16 + 32 > j * SELW))
        pv += pgrp[t];
    int cur = s / SELW;
    if (j > cur) pv = -BIGF;
    if (j == 0 || j == cur) pv = BIGF;
    float work = pv;
    for (int i = 0; i < TOPN; ++i) {
      float cv = work; int ci = j;
      #pragma unroll
      for (int d = 16; d >= 1; d >>= 1) {          // argmax, ties -> low idx
        float ov = __shfl_xor(cv, d, 32);
        int   oi = __shfl_xor(ci, d, 32);
        if (ov > cv || (ov == cv && oi < ci)) { cv = ov; ci = oi; }
      }
      if (j == ci) work = -3.0e38f;                // remove winner
      if (lane == 0) selidx[i] = (cv <= -0.5f * BIGF) ? 0 : ci;
    }
    __syncthreads();
    if (lane < TOPN) {
      int me = selidx[lane], dupf = 0;
      for (int p2 = 0; p2 < lane; ++p2) dupf |= (selidx[p2] == me);
      seldup[lane] = dupf;
    }
    __syncthreads();
  }

  // ---------------- selected attention (8 blocks x 64 tokens) -------------
  Flash st;
  st.reset();
  for (int rank = 0; rank < TOPN; ++rank) {
    if (seldup[rank]) continue;                    // dup blocks fully masked
    int blk = selidx[rank];
    attn_group32(st, qa, kh, vh, blk * SELW,      0, s, S_LEN - 1, vstage, pstage, lane);
    attn_group32(st, qa, kh, vh, blk * SELW + 32, 0, s, S_LEN - 1, vstage, pstage, lane);
  }
  #pragma unroll
  for (int nc = 0; nc < 8; ++nc)
    #pragma unroll
    for (int r = 0; r < 8; ++r) {
      int h = r + half * 8;
      outbuf[h * DV + nc * 16 + ln] +=
          gates[h * 3 + 1] * st.acc[nc][r] / fmaxf(st.l[r], 1e-20f);
    }

  // ---------------- sliding-window attention -------------------------------
  st.reset();
  int lo = (s >= WIN) ? s - WIN : 0;               // t in [s-512, s]
  for (int tb = lo & ~31; tb <= s; tb += 32)
    attn_group32(st, qa, kh, vh, tb, lo, s, S_LEN - 1, vstage, pstage, lane);
  #pragma unroll
  for (int nc = 0; nc < 8; ++nc)
    #pragma unroll
    for (int r = 0; r < 8; ++r) {
      int h = r + half * 8;
      outbuf[h * DV + nc * 16 + ln] +=
          gates[h * 3 + 2] * st.acc[nc][r] / fmaxf(st.l[r], 1e-20f);
    }

  __syncthreads();
  // vectorized output copy (8KB aligned global offset, 16B aligned LDS)
  {
    float4* outp4 = (float4*)(out + (size_t)s * HQ * DV);
    const float4* ob4 = (const float4*)outbuf;
    for (int i = lane; i < HQ * DV / 4; i += 32) outp4[i] = ob4[i];
  }
}

extern "C" void kernel_launch(void* const* d_in, const int* in_sizes, int n_in,
                              void* d_out, int out_size, void* d_ws, size_t ws_size,
                              hipStream_t stream)
{
  const float* q  = (const float*)d_in[0];
  const float* k  = (const float*)d_in[1];
  const float* v  = (const float*)d_in[2];
  const float* wk = (const float*)d_in[3];
  const float* wv = (const float*)d_in[4];
  const float* gw = (const float*)d_in[5];
  const float* gb = (const float*)d_in[6];
  float* out = (float*)d_out;

  // ws layout (all 32B-aligned): kh, vh, cmp_k, cmp_v in f16 (~1.1 MB total)
  _Float16* kh    = (_Float16*)d_ws;
  _Float16* vh    = kh + (size_t)S_LEN * DK;
  _Float16* cmpkh = vh + (size_t)S_LEN * DK;
  _Float16* cmpvh = cmpkh + (size_t)T_CMP * DK;

  hipLaunchKernelGGL(nsa_prep, dim3((S_LEN * DK + 255) / 256), dim3(256), 0, stream,
                     k, v, wk, wv, kh, vh, cmpkh, cmpvh);
  hipLaunchKernelGGL(nsa_main, dim3(S_LEN), dim3(32), 0, stream,
                     q, kh, vh, cmpkh, cmpvh, gw, gb, out);
}
